// NodeModel_22728966930783
// MI455X (gfx1250) — matrix operand order repaired and checked by reading, not compiled, not run on hardware
//
#include <hip/hip_runtime.h>
#include <hip/hip_bf16.h>

#define DH   128
#define DF   6
#define K1   288   // 262 (=2*DH+DF) padded up to a multiple of 32
#define HSTR 296   // LDS row stride in bf16 elements (padded vs 288)

typedef __attribute__((ext_vector_type(16))) __bf16 v16bf;
typedef __attribute__((ext_vector_type(8)))  float  v8f;
typedef __attribute__((ext_vector_type(4)))  float  v4f;

union Frag { v16bf v; uint4 q[2]; };
union PackB4 { __bf16 h[4]; uint2 u; };

// ---------------- zero scratch (sums + counts, contiguous) ----------------
__global__ void k_zero(float* __restrict__ p, long n) {
  long i = (long)blockIdx.x * 256 + threadIdx.x;
  if (i < n) p[i] = 0.0f;
}

// ------------- weights: transpose to [N][K] row-major, bf16, pad ----------
__global__ void k_prep(const float* __restrict__ W1, const float* __restrict__ W2,
                       const float* __restrict__ W3, __bf16* __restrict__ w1t,
                       __bf16* __restrict__ w2t, __bf16* __restrict__ w3t) {
  int i = blockIdx.x * 256 + threadIdx.x;
  if (i < 128 * K1) {
    int n = i / K1, k = i - n * K1;
    w1t[i] = (__bf16)((k < 2 * DH + DF) ? W1[k * 128 + n] : 0.0f);
  }
  int j = i - 128 * K1;
  if (j >= 0 && j < 128 * 128) {
    int n = j >> 7, k = j & 127;
    w2t[j] = (__bf16)W2[k * 128 + n];
    w3t[j] = (__bf16)W3[k * 128 + n];
  }
}

// ---------------- scatter-add: one wave per edge, float4/lane -------------
// edge_attr is an 819MB one-shot stream -> non-temporal loads so the 51MB
// atomic accumulator stays resident in the 192MB L2.
__global__ void k_scatter(const float* __restrict__ ea, const long long* __restrict__ ei,
                          float* __restrict__ sums, float* __restrict__ counts, int E) {
  int e    = (blockIdx.x * 256 + threadIdx.x) >> 5;
  int lane = threadIdx.x & 31;
  if (e >= E) return;
  long long d = ei[(long long)E + e];                 // edge_index[1][e]
  v4f v = __builtin_nontemporal_load((const v4f*)(ea + (long long)e * DH + lane * 4));
  float* s = sums + d * DH + lane * 4;
  atomicAdd(s + 0, v.x); atomicAdd(s + 1, v.y);
  atomicAdd(s + 2, v.z); atomicAdd(s + 3, v.w);
  if (lane == 0) atomicAdd(counts + d, 1.0f);
}

// --------------------------- fused 3-layer MLP ----------------------------
__device__ __forceinline__ void load_w(const __bf16* __restrict__ g, __bf16* wbuf, int KC) {
  int cpr  = KC >> 3;          // 16-byte chunks per row
  int nvec = 128 * cpr;
  for (int i = threadIdx.x; i < nvec; i += 256) {
    int r = i / cpr, c = i - r * cpr;
    *(uint4*)(wbuf + r * HSTR + c * 8) = *(const uint4*)(g + r * KC + c * 8);
  }
}

__device__ __forceinline__ void gemm_tile(const __bf16* hbuf, const __bf16* wbuf,
                                          const float* __restrict__ bias,
                                          v8f acc[8], int KC, int wave, int lane) {
  int lx = lane & 15, lh = lane >> 4;
#pragma unroll
  for (int tn = 0; tn < 8; ++tn) {
    float bv = bias[tn * 16 + lx];                    // bias depends only on N column
#pragma unroll
    for (int i = 0; i < 8; ++i) acc[tn][i] = bv;
  }
  const __bf16* arow = hbuf + (wave * 16 + lx) * HSTR;
  for (int kk = 0; kk < KC; kk += 32) {
    Frag a;                                           // A: lanes 0-15 K[0..7]+[16..23], 16-31 K[8..15]+[24..31]
    a.q[0] = *(const uint4*)(arow + kk + lh * 8);
    a.q[1] = *(const uint4*)(arow + kk + lh * 8 + 16);
#pragma unroll
    for (int tn = 0; tn < 8; ++tn) {
      const __bf16* brow = wbuf + (tn * 16 + lx) * HSTR;
      Frag b;                                         // B: lane = column, lanes 0-15 K[0..7]+[16..23], 16-31 K[8..15]+[24..31]
      b.q[0] = *(const uint4*)(brow + kk + lh * 8);
      b.q[1] = *(const uint4*)(brow + kk + lh * 8 + 16);
      acc[tn] = __builtin_amdgcn_wmma_f32_16x16x32_bf16(
          false, a.v, false, b.v, (short)0, acc[tn], false, false);
    }
  }
}

__device__ __forceinline__ void pack4(__bf16* dst, v4f v) {
  PackB4 p;
  p.h[0] = (__bf16)v.x; p.h[1] = (__bf16)v.y;
  p.h[2] = (__bf16)v.z; p.h[3] = (__bf16)v.w;
  *(uint2*)dst = p.u;                                 // ds_store_b64
}

__global__ __launch_bounds__(256) void k_mlp(
    const float* __restrict__ x, const float* __restrict__ f,
    const float* __restrict__ sums, const float* __restrict__ counts,
    const __bf16* __restrict__ w1t, const __bf16* __restrict__ w2t,
    const __bf16* __restrict__ w3t,
    const float* __restrict__ b1, const float* __restrict__ b2,
    const float* __restrict__ b3, float* __restrict__ out, int N) {
  extern __shared__ __align__(16) char smem[];
  __bf16* hbuf = (__bf16*)smem;            // [128][HSTR] activations (bf16)
  __bf16* wbuf = hbuf + 128 * HSTR;        // [128][HSTR] current weight tile (W^T)
  int lane = threadIdx.x & 31;
  int wave = threadIdx.x >> 5;
  int lx = lane & 15, lh = lane >> 4;
  long nodeBase = (long)blockIdx.x * 128;

  // ---- build A = [x | mean-agg | f | 0pad] in bf16; wave-per-16-rows,
  //      float4 loads (x non-temporal: read-once stream), b64 LDS stores ----
  for (int rr = 0; rr < 16; ++rr) {
    int r = wave * 16 + rr;
    long node  = nodeBase + r;
    bool valid = node < N;
    __bf16* hrow = hbuf + r * HSTR;
    v4f z = {0.0f, 0.0f, 0.0f, 0.0f};
    float invc = valid ? (1.0f / fmaxf(counts[node], 1.0f)) : 0.0f;
    v4f vx = valid ? __builtin_nontemporal_load((const v4f*)(x + node * DH + lane * 4)) : z;
    pack4(hrow + lane * 4, vx);
    v4f vs = valid ? *(const v4f*)(sums + node * DH + lane * 4) : z;
    vs.x *= invc; vs.y *= invc; vs.z *= invc; vs.w *= invc;
    pack4(hrow + DH + lane * 4, vs);
    float vf = (valid && lane < DF) ? f[node * DF + lane] : 0.0f;
    hrow[2 * DH + lane] = (__bf16)vf;                 // cols 256..287 (pad -> 0)
  }
  load_w(w1t, wbuf, K1);
  __syncthreads();

  v8f acc[8];
  // ---- layer 1: [128 x 288] @ [288 x 128] + b1, SiLU ----
  gemm_tile(hbuf, wbuf, b1, acc, K1, wave, lane);
#pragma unroll
  for (int tn = 0; tn < 8; ++tn)
#pragma unroll
    for (int i = 0; i < 8; ++i) {
      float v = acc[tn][i];
      v = v / (1.0f + __expf(-v));                   // silu
      hbuf[(wave * 16 + lh * 8 + i) * HSTR + tn * 16 + lx] = (__bf16)v;  // own strip only
    }
  __syncthreads();
  load_w(w2t, wbuf, 128);
  __syncthreads();

  // ---- layer 2: [128 x 128] @ [128 x 128] + b2, SiLU ----
  gemm_tile(hbuf, wbuf, b2, acc, 128, wave, lane);
#pragma unroll
  for (int tn = 0; tn < 8; ++tn)
#pragma unroll
    for (int i = 0; i < 8; ++i) {
      float v = acc[tn][i];
      v = v / (1.0f + __expf(-v));
      hbuf[(wave * 16 + lh * 8 + i) * HSTR + tn * 16 + lx] = (__bf16)v;
    }
  __syncthreads();
  load_w(w3t, wbuf, 128);
  __syncthreads();

  // ---- layer 3: [128 x 128] @ [128 x 128] + b3, no activation ----
  gemm_tile(hbuf, wbuf, b3, acc, 128, wave, lane);
#pragma unroll
  for (int tn = 0; tn < 8; ++tn)
#pragma unroll
    for (int i = 0; i < 8; ++i) {
      long node = nodeBase + wave * 16 + lh * 8 + i;
      if (node < N)
        __builtin_nontemporal_store(acc[tn][i], out + node * DH + tn * 16 + lx);
    }
}

// --------------------------------------------------------------------------
extern "C" void kernel_launch(void* const* d_in, const int* in_sizes, int n_in,
                              void* d_out, int out_size, void* d_ws, size_t ws_size,
                              hipStream_t stream) {
  const float*     x  = (const float*)d_in[0];
  const long long* ei = (const long long*)d_in[1];   // int64 [2, E]
  const float*     ea = (const float*)d_in[2];
  const float*     f  = (const float*)d_in[3];
  const float*     W1 = (const float*)d_in[4];
  const float*     b1 = (const float*)d_in[5];
  const float*     W2 = (const float*)d_in[6];
  const float*     b2 = (const float*)d_in[7];
  const float*     W3 = (const float*)d_in[8];
  const float*     b3 = (const float*)d_in[9];
  float* out = (float*)d_out;

  int N = in_sizes[0] / DH;
  int E = in_sizes[2] / DH;

  char*  ws     = (char*)d_ws;
  float* sums   = (float*)ws;                        // [N][128]
  float* counts = sums + (size_t)N * DH;             // [N] (contiguous -> one zero pass)
  size_t woff   = (((size_t)N * (DH + 1) * sizeof(float)) + 255) & ~(size_t)255;
  __bf16* w1t = (__bf16*)(ws + woff);                // [128][288]
  __bf16* w2t = w1t + 128 * K1;                      // [128][128]
  __bf16* w3t = w2t + 128 * 128;                     // [128][128]

  long ztot = (long)N * (DH + 1);
  k_zero<<<(int)((ztot + 255) / 256), 256, 0, stream>>>(sums, ztot);

  int ptot = 128 * K1 + 128 * 128;
  k_prep<<<(ptot + 255) / 256, 256, 0, stream>>>(W1, W2, W3, w1t, w2t, w3t);

  k_scatter<<<(E + 7) / 8, 256, 0, stream>>>(ea, ei, sums, counts, E);

  size_t smem = (size_t)2 * 128 * HSTR * sizeof(__bf16);   // 151,552 B < 320 KB/WGP
  k_mlp<<<(N + 127) / 128, 256, smem, stream>>>(x, f, sums, counts,
                                                w1t, w2t, w3t, b1, b2, b3, out, N);
}
